// S4Encoder_78709570666711
// MI455X (gfx1250) — matrix operand order repaired and checked by reading, not compiled, not run on hardware
//
#include <hip/hip_runtime.h>
#include <hip/hip_bf16.h>

typedef __bf16 bf16_t;
typedef __bf16 v16bf __attribute__((ext_vector_type(16)));
typedef float  v8f   __attribute__((ext_vector_type(8)));
typedef unsigned int v4u __attribute__((ext_vector_type(4)));
typedef int    v8i   __attribute__((ext_vector_type(8)));
typedef int    v4i   __attribute__((ext_vector_type(4)));

constexpr int B  = 16;
constexpr int L  = 2048;
constexpr int H  = 512;
constexpr int NH = 32;
constexpr int NL = 4;
constexpr int O2 = 2 * H;   // 1024 rows of the 1x1 conv

// ---------------------------------------------------------------------------
// 1) Precompute per-(layer,h,n) discretized SSM params:
//    lambda = exp(dt*A),  Ct = 2 * C * (exp(dt*A)-1)/A   (complex)
// ---------------------------------------------------------------------------
__global__ void s4d_precompute_kernel(const float* __restrict__ log_dt,      // (NL,H)
                                      const float* __restrict__ Cri,         // (NL,H,NH,2)
                                      const float* __restrict__ log_A_real,  // (NL,H,NH)
                                      const float* __restrict__ A_imag,      // (NL,H,NH)
                                      float4* __restrict__ params)           // (NL,H,NH)
{
    int idx = blockIdx.x * blockDim.x + threadIdx.x;
    if (idx >= NL * H * NH) return;
    int ih = idx / NH;  // layer*H + h
    float dt  = expf(log_dt[ih]);
    float Are = -expf(log_A_real[idx]);
    float Aim = A_imag[idx];
    float er  = expf(Are * dt);
    float lam_re = er * cosf(Aim * dt);
    float lam_im = er * sinf(Aim * dt);
    float num_re = lam_re - 1.0f, num_im = lam_im;
    float inv_den = 1.0f / (Are * Are + Aim * Aim);
    float q_re = (num_re * Are + num_im * Aim) * inv_den;
    float q_im = (num_im * Are - num_re * Aim) * inv_den;
    float Cre = Cri[2 * idx], Cim = Cri[2 * idx + 1];
    params[idx] = make_float4(lam_re, lam_im,
                              2.0f * (Cre * q_re - Cim * q_im),
                              2.0f * (Cre * q_im + Cim * q_re));
}

// ---------------------------------------------------------------------------
// 2) Encoder: h[b,l,c] = x[b,l]*enc_w[c] + enc_b[c]   (B,L,H layout, f32)
// ---------------------------------------------------------------------------
__global__ void encoder_kernel(const float* __restrict__ x,
                               const float* __restrict__ enc_w,
                               const float* __restrict__ enc_b,
                               float* __restrict__ hbuf)
{
    int idx = blockIdx.x * blockDim.x + threadIdx.x;
    if (idx >= B * L * H) return;
    int c  = idx & (H - 1);
    int bl = idx >> 9;          // /H
    hbuf[idx] = x[bl] * enc_w[c] + enc_b[c];
}

// ---------------------------------------------------------------------------
// 3) out_w f32 -> bf16 (all layers at once)
// ---------------------------------------------------------------------------
__global__ void w_to_bf16_kernel(const float* __restrict__ w,
                                 bf16_t* __restrict__ wbf, int n)
{
    int idx = blockIdx.x * blockDim.x + threadIdx.x;
    if (idx < n) wbf[idx] = (bf16_t)w[idx];
}

// ---------------------------------------------------------------------------
// 4) Diagonal SSM scan: one wave32 per (b,h); lane n owns complex state n.
//    y = conv(u) + D*u, then exact GELU, stored bf16 in (B,L,H).
// ---------------------------------------------------------------------------
__global__ void __launch_bounds__(256)
s4d_scan_kernel(const float* __restrict__ hin,      // (B,L,H) f32
                const float4* __restrict__ params,  // layer slice (H,NH)
                const float* __restrict__ Dvec,     // layer slice (H)
                bf16_t* __restrict__ ybf)           // (B,L,H) bf16
{
    int wave = (blockIdx.x * blockDim.x + threadIdx.x) >> 5;
    int lane = threadIdx.x & 31;
    int b  = wave / H;
    int hh = wave - b * H;

    float4 p = params[hh * NH + lane];   // lam_re, lam_im, ct_re, ct_im
    float d  = Dvec[hh];
    float s_re = 0.0f, s_im = 0.0f;

    const float* up = hin + (size_t)b * L * H + hh;
    bf16_t*      yp = ybf + (size_t)b * L * H + hh;

    for (int l = 0; l < L; ++l) {
        float u  = up[(size_t)l * H];
        float nr = fmaf(p.x, s_re, fmaf(-p.y, s_im, u));
        float ni = fmaf(p.x, s_im, p.y * s_re);
        s_re = nr; s_im = ni;
        float c = p.z * s_re - p.w * s_im;     // Re(Ct * s), factor 2 folded in
        #pragma unroll
        for (int off = 16; off > 0; off >>= 1)
            c += __shfl_xor(c, off, 32);       // wave32 butterfly reduce
        if (lane == 0) {
            float y = c + u * d;
            y = 0.5f * y * (1.0f + erff(y * 0.70710678118654752f));  // exact GELU
            yp[(size_t)l * H] = (bf16_t)y;
        }
    }
}

// ---------------------------------------------------------------------------
// 5) Fused: v = W(1024x512) * y(512 x 16cols) via v_wmma_f32_16x16x32_bf16.
//    The 16x512 bf16 y-tile is staged into LDS by the Tensor Data Mover
//    (tensor_load_to_lds, waited with s_wait_tensorcnt), aliasing the first
//    16KB of the 64KB accumulator buffer.  Then bias+GLU+residual+LayerNorm,
//    in-place update of hbuf.  Block = 256 threads = 8 waves x 8 row-tiles.
// ---------------------------------------------------------------------------
__global__ void __launch_bounds__(256)
gemm_glu_ln_kernel(const bf16_t* __restrict__ Wbf,   // (2H,H) layer slice, row-major
                   const bf16_t* __restrict__ ybf,   // (B*L, H) bf16
                   const float*  __restrict__ bias,  // (2H) layer slice
                   const float*  __restrict__ lng,   // (H)
                   const float*  __restrict__ lnb,   // (H)
                   float* __restrict__ hbuf)         // (B*L, H) f32, in/out
{
    __shared__ float vbuf[O2 * 16];        // 64 KB; first 16 KB doubles as y tile
    bf16_t* ytile = (bf16_t*)vbuf;         // 16 columns x 512 K-values, bf16

    const int tid  = threadIdx.x;
    const int lane = tid & 31;
    const int w    = tid >> 5;             // wave 0..7
    const int g    = lane >> 4;            // half-wave 0/1
    const int lm   = lane & 15;
    const int nc0  = blockIdx.x * 16;      // first global column (b*L + l)

    // ---- TDM: one async tensor load per block: 16 rows x 512 elems of bf16 ----
    if (w == 0) {
        unsigned long long ga = (unsigned long long)(ybf + (size_t)nc0 * H);
        // D# group 0: count=1, lds_addr=0, global_addr, type=2
        v4u g0 = { 1u, 0u,
                   (unsigned)(ga & 0xffffffffu),
                   (unsigned)(ga >> 32) | (2u << 30) };
        // D# group 1: data_size=2B, tensor_dim0=H, tensor_dim1=B*L,
        //             tile_dim0=H, tile_dim1=16, tensor_dim0_stride=H
        v8i g1 = { (int)(1u << 16),                               // data_size code 1
                   (int)((unsigned)H << 16),                      // tensor_dim0[15:0]
                   (int)(((unsigned)(B * L) & 0xffffu) << 16),    // tensor_dim1[15:0]
                   (int)((unsigned)H << 16),                      // tile_dim0
                   16,                                            // tile_dim1
                   H,                                             // dim0_stride lo32
                   0, 0 };
        v4i g2 = { 0, 0, 0, 0 };
        v4i g3 = { 0, 0, 0, 0 };
        v8i g4 = { 0, 0, 0, 0, 0, 0, 0, 0 };
        __builtin_amdgcn_tensor_load_to_lds(g0, g1, g2, g3, g4, 0);
        __builtin_amdgcn_s_wait_tensorcnt(0);
    }
    __syncthreads();

    // ---------------- GEMM ----------------
    v8f acc[8];
    #pragma unroll
    for (int m = 0; m < 8; ++m)
        #pragma unroll
        for (int r = 0; r < 8; ++r) acc[m][r] = 0.0f;

    const bf16_t* myy = ytile + lm * H;    // this lane's column in LDS

    for (int kt = 0; kt < H / 32; ++kt) {
        // B fragment (32x16 bf16) from LDS: 16 contiguous bf16 per lane
        v16bf bfrag;
        {
            int kbase = kt * 32 + 16 * g;
            #pragma unroll
            for (int e = 0; e < 16; ++e) bfrag[e] = myy[kbase + e];
        }
        // Batch-load all 8 A fragments first so the WMMAs below can overlap
        // the tail loads (staged s_wait_loadcnt instead of wait-0 per WMMA).
        v16bf afr[8];
        #pragma unroll
        for (int m = 0; m < 8; ++m) {
            int row = (w * 8 + m) * 16 + lm;           // output channel o
            const bf16_t* arow = Wbf + (size_t)row * H;
            #pragma unroll
            for (int j = 0; j < 8; ++j) {
                int kb = kt * 32 + 16 * g + 2 * j;
                afr[m][2 * j]     = arow[kb];
                afr[m][2 * j + 1] = arow[kb + 1];
            }
        }
        #pragma unroll
        for (int m = 0; m < 8; ++m)
            acc[m] = __builtin_amdgcn_wmma_f32_16x16x32_bf16(
                         false, afr[m], false, bfrag, (short)0, acc[m],
                         false, false);
    }

    __syncthreads();   // everyone done reading ytile before vbuf overwrites it

    // D layout: VGPR r -> row r + 8*g, col = lane&15
    #pragma unroll
    for (int m = 0; m < 8; ++m)
        #pragma unroll
        for (int r = 0; r < 8; ++r) {
            int row = (w * 8 + m) * 16 + r + 8 * g;
            vbuf[row * 16 + lm] = acc[m][r];
        }
    __syncthreads();

    // ---------------- GLU + bias + residual ----------------
    for (int e = tid; e < H * 16; e += 256) {
        int o = e >> 4, n = e & 15;
        float a  = vbuf[o * 16 + n] + bias[o];
        float gt = vbuf[(o + H) * 16 + n] + bias[o + H];
        float z  = a * (1.0f / (1.0f + expf(-gt)));
        z += hbuf[(size_t)(nc0 + n) * H + o];
        vbuf[o * 16 + n] = z;
    }
    __syncthreads();

    // ---------------- LayerNorm over H per column ----------------
    float* sred = vbuf + H * 16;   // reuse upper half (rows 512..1023)
    {
        int col = tid & 15, sub = tid >> 4;   // 16 partials per column
        float s = 0.0f, q = 0.0f;
        for (int r = sub * 32; r < sub * 32 + 32; ++r) {
            float z = vbuf[r * 16 + col];
            s += z; q += z * z;
        }
        sred[col * 16 + sub]       = s;
        sred[256 + col * 16 + sub] = q;
    }
    __syncthreads();
    if (tid < 16) {
        float s = 0.0f, q = 0.0f;
        for (int sub = 0; sub < 16; ++sub) {
            s += sred[tid * 16 + sub];
            q += sred[256 + tid * 16 + sub];
        }
        float mean = s * (1.0f / H);
        float var  = q * (1.0f / H) - mean * mean;
        sred[512 + tid] = mean;
        sred[528 + tid] = rsqrtf(var + 1e-5f);
    }
    __syncthreads();
    for (int e = tid; e < H * 16; e += 256) {
        int o = e >> 4, n = e & 15;
        float z = vbuf[o * 16 + n];
        hbuf[(size_t)(nc0 + n) * H + o] =
            (z - sred[512 + n]) * sred[528 + n] * lng[o] + lnb[o];
    }
}

// ---------------------------------------------------------------------------
// 6) Mean pool over L  -> feats (B,H)
// ---------------------------------------------------------------------------
__global__ void pool_kernel(const float* __restrict__ hbuf,
                            float* __restrict__ feats)
{
    int idx = blockIdx.x * blockDim.x + threadIdx.x;
    if (idx >= B * H) return;
    int b = idx / H, c = idx - (idx / H) * H;
    const float* p = hbuf + (size_t)b * L * H + c;
    float s = 0.0f;
    for (int l = 0; l < L; ++l) s += p[(size_t)l * H];
    feats[idx] = s * (1.0f / L);
}

// ---------------------------------------------------------------------------
// 7) Decoder: out[b,o] = feats[b,:] @ dec_w[:,o] + dec_b[o]   (tiny)
// ---------------------------------------------------------------------------
__global__ void decoder_kernel(const float* __restrict__ feats,
                               const float* __restrict__ dec_w,
                               const float* __restrict__ dec_b,
                               float* __restrict__ out)
{
    int idx = blockIdx.x * blockDim.x + threadIdx.x;
    if (idx >= B * H) return;
    int b = idx / H, o = idx - (idx / H) * H;
    const float* f = feats + (size_t)b * H;
    float s = dec_b[o];
    for (int h = 0; h < H; ++h) s = fmaf(f[h], dec_w[(size_t)h * H + o], s);
    out[idx] = s;
}

// ---------------------------------------------------------------------------
extern "C" void kernel_launch(void* const* d_in, const int* in_sizes, int n_in,
                              void* d_out, int out_size, void* d_ws, size_t ws_size,
                              hipStream_t stream)
{
    const float* x      = (const float*)d_in[0];
    const float* enc_w  = (const float*)d_in[1];
    const float* enc_b  = (const float*)d_in[2];
    const float* log_dt = (const float*)d_in[3];
    const float* Cri    = (const float*)d_in[4];
    const float* logA   = (const float*)d_in[5];
    const float* Aim    = (const float*)d_in[6];
    const float* Dv     = (const float*)d_in[7];
    const float* out_w  = (const float*)d_in[8];
    const float* out_b  = (const float*)d_in[9];
    const float* ln_g   = (const float*)d_in[10];
    const float* ln_b   = (const float*)d_in[11];
    const float* dec_w  = (const float*)d_in[12];
    const float* dec_b  = (const float*)d_in[13];
    float* out = (float*)d_out;

    char* ws = (char*)d_ws;
    size_t off = 0;
    float4* params = (float4*)(ws + off); off += (size_t)NL * H * NH * sizeof(float4);
    float*  hbuf   = (float*)(ws + off);  off += (size_t)B * L * H * sizeof(float);
    bf16_t* ybf    = (bf16_t*)(ws + off); off += (size_t)B * L * H * sizeof(bf16_t);
    bf16_t* wbf    = (bf16_t*)(ws + off); off += (size_t)NL * O2 * H * sizeof(bf16_t);
    float*  feats  = (float*)(ws + off);  off += (size_t)B * H * sizeof(float);
    (void)ws_size; (void)in_sizes; (void)n_in; (void)out_size;

    s4d_precompute_kernel<<<(NL * H * NH) / 256, 256, 0, stream>>>(
        log_dt, Cri, logA, Aim, params);
    encoder_kernel<<<(B * L * H) / 256, 256, 0, stream>>>(x, enc_w, enc_b, hbuf);
    w_to_bf16_kernel<<<(NL * O2 * H) / 256, 256, 0, stream>>>(out_w, wbf, NL * O2 * H);

    for (int i = 0; i < NL; ++i) {
        s4d_scan_kernel<<<(B * H) / 8, 256, 0, stream>>>(
            hbuf, params + (size_t)i * H * NH, Dv + (size_t)i * H, ybf);
        gemm_glu_ln_kernel<<<(B * L) / 16, 256, 0, stream>>>(
            wbf + (size_t)i * O2 * H, ybf, out_b + (size_t)i * O2,
            ln_g + (size_t)i * H, ln_b + (size_t)i * H, hbuf);
    }

    pool_kernel<<<(B * H) / 256, 256, 0, stream>>>(hbuf, feats);
    decoder_kernel<<<(B * H) / 256, 256, 0, stream>>>(feats, dec_w, dec_b, out);
}
